// TensorialDecoder_10187662426204
// MI455X (gfx1250) — compile-verified, matrix operand order checked by reading, not run on hardware
//
#include <hip/hip_runtime.h>
#include <hip/hip_bf16.h>
#include <math.h>

typedef __attribute__((ext_vector_type(16))) _Float16 v16h;
typedef __attribute__((ext_vector_type(8)))  _Float16 v8h;
typedef __attribute__((ext_vector_type(8)))  float    v8f;
typedef __attribute__((ext_vector_type(2), aligned(4))) float v2f_u;  // align-4 8B load

#define C_FEAT 32
#define RES 512

__device__ __forceinline__ void wave_lds_fence() {
    // Wave-local LDS RAW ordering: DS ops are in-order per wave; the asm's
    // memory clobber prevents compiler reordering, s_wait_dscnt drains them.
    asm volatile("s_wait_dscnt 0" ::: "memory");
    __builtin_amdgcn_wave_barrier();
}

// Load one 16-bit A/B fragment row (v16h) from LDS: two 16-byte chunks at
// half-offsets g0 and g0+16 within the row (g0 = 8 for lanes 16..31).
__device__ __forceinline__ v16h load_frag(const _Float16* row, int g0) {
    v8h lo = *(const v8h*)(row + g0);
    v8h hi = *(const v8h*)(row + g0 + 16);
    return __builtin_shufflevector(lo, hi, 0,1,2,3,4,5,6,7,8,9,10,11,12,13,14,15);
}

__device__ __forceinline__ float fast_sigmoid(float x) {
    // v_exp_f32 + v_add + v_rcp_f32 (no IEEE divide expansion)
    return __builtin_amdgcn_rcpf(1.0f + __expf(-x));
}
__device__ __forceinline__ float fast_silu(float x) {
    return x * fast_sigmoid(x);
}

__global__ __launch_bounds__(256)
void tensorial_decoder_kernel(const float* __restrict__ code,
                              const float* __restrict__ pts,
                              const float* __restrict__ dirs,
                              const float* __restrict__ w_base,
                              const float* __restrict__ b_base,
                              const float* __restrict__ w_density,
                              const float* __restrict__ b_density,
                              const float* __restrict__ w_dir,
                              const float* __restrict__ b_dir,
                              const float* __restrict__ w_color,
                              const float* __restrict__ b_color,
                              const float* __restrict__ w_color_wodir,
                              const float* __restrict__ b_color_wodir,
                              float* __restrict__ out,
                              int npts) {
    // ---- LDS ----
    __shared__ _Float16 sWB[64 * 32];      // w_base  [n][k]
    __shared__ _Float16 sWD[64 * 32];      // w_dir   [n][k], k>=9 zero
    __shared__ _Float16 sB1[16 * 64];      // head1 [n][k]: n0=density, n1..3=wodir
    __shared__ _Float16 sB2[16 * 64];      // head2 [n][k]: n1..3=color
    __shared__ _Float16 sA  [8][16 * 32];  // per-wave point_code tile (f16)
    __shared__ _Float16 sSH [8][16 * 32];  // per-wave SH tile (padded K=32)
    __shared__ _Float16 sAct1[8][16 * 64]; // silu(base_x)
    __shared__ _Float16 sAct2[8][16 * 64]; // silu(base_x + sh@w_dir.T + b_dir)

    const int tid = threadIdx.x;

    // ---- Cooperative weight staging (once per block) ----
    for (int i = tid; i < 64 * 32; i += 256)
        sWB[i] = (_Float16)w_base[i];                        // (64,32) row-major == [n][k]
    for (int i = tid; i < 64 * 32; i += 256) {
        int n = i >> 5, k = i & 31;
        sWD[i] = (_Float16)((k < 9) ? w_dir[n * 9 + k] : 0.0f);
    }
    for (int i = tid; i < 16 * 64; i += 256) {
        int n = i >> 6, k = i & 63;
        float v = 0.0f;
        if (n == 0) v = w_density[k];
        else if (n <= 3) v = w_color_wodir[(n - 1) * 64 + k];
        sB1[i] = (_Float16)v;
    }
    for (int i = tid; i < 16 * 64; i += 256) {
        int n = i >> 6, k = i & 63;
        float v = (n >= 1 && n <= 3) ? w_color[(n - 1) * 64 + k] : 0.0f;
        sB2[i] = (_Float16)v;
    }
    __syncthreads();

    const int wave = tid >> 5;
    const int lane = tid & 31;
    const int m    = lane & 15;        // point row within tile / fragment column
    const int hi   = lane >> 4;        // K-half selector
    const int g0   = hi * 8;           // half-offset within 32-half row
    const int tile = blockIdx.x * 8 + wave;
    const int p0   = tile * 16;
    if (p0 >= npts) return;
    const int p = p0 + m;              // this lane's sampling point

    // ---- Tri-plane bilinear sampling: point_code[c] = c0*c1*c2 ----
    {
        const float px = pts[3 * p + 0];
        const float py = pts[3 * p + 1];
        const float pz = pts[3 * p + 2];
        const float ax[3] = {px, py, pz};
        const int pa[3] = {0, 1, 2};
        const int pb[3] = {1, 2, 0};

        // Clamp x0,y0 <= RES-2 and fold the border into wx,wy -> exact same
        // bilinear result, but x1 == x0+1 always => float2 row loads.
        int   O0[3], O1[3];
        float WX[3], WY[3];
#pragma unroll
        for (int k = 0; k < 3; ++k) {
            float fx = (ax[pa[k]] + 1.0f) * 0.5f * (float)(RES - 1);
            float fy = (ax[pb[k]] + 1.0f) * 0.5f * (float)(RES - 1);
            fx = fminf(fmaxf(fx, 0.0f), (float)(RES - 1));
            fy = fminf(fmaxf(fy, 0.0f), (float)(RES - 1));
            int x0 = min((int)fx, RES - 2);
            int y0 = min((int)fy, RES - 2);
            WX[k] = fx - (float)x0;
            WY[k] = fy - (float)y0;
            O0[k] = y0 * RES + x0;
            O1[k] = O0[k] + RES;
        }

        const int chbase = hi * 16;    // this lane covers 16 of 32 channels
        float prod[16];
#pragma unroll
        for (int c = 0; c < 16; ++c) prod[c] = 1.0f;

        for (int k = 0; k < 3; ++k) {
            const float* fb = code + (((size_t)(k * C_FEAT + chbase)) << 18);
            const int   o0 = O0[k], o1 = O1[k];
            const float wx = WX[k], wy = WY[k];
#pragma unroll
            for (int c = 0; c < 16; ++c) {
                const float* f = fb + ((size_t)c << 18);
                v2f_u r0 = *(const v2f_u*)(f + o0);   // f00, f01
                v2f_u r1 = *(const v2f_u*)(f + o1);   // f10, f11
                float top = r0.x + wx * (r0.y - r0.x);
                float bot = r1.x + wx * (r1.y - r1.x);
                prod[c] *= top + wy * (bot - top);
            }
        }

        v8h h0, h1v;
#pragma unroll
        for (int c = 0; c < 8; ++c) { h0[c] = (_Float16)prod[c]; h1v[c] = (_Float16)prod[c + 8]; }
        *(v8h*)&sA[wave][m * 32 + chbase]     = h0;
        *(v8h*)&sA[wave][m * 32 + chbase + 8] = h1v;
    }

    // ---- SH encode (9 values, K padded to 32) ----
    if (hi == 0) {
        const float dx = dirs[3 * p + 0];
        const float dy = dirs[3 * p + 1];
        const float dz = dirs[3 * p + 2];
        float s[16];
        s[0] =  0.28209479177387814f;
        s[1] = -0.4886025119029199f * dy;
        s[2] =  0.4886025119029199f * dz;
        s[3] = -0.4886025119029199f * dx;
        s[4] =  1.0925484305920792f * dx * dy;
        s[5] = -1.0925484305920792f * dy * dz;
        s[6] =  0.31539156525252005f * (2.0f * dz * dz - dx * dx - dy * dy);
        s[7] = -1.0925484305920792f * dx * dz;
        s[8] =  0.5462742152960396f * (dx * dx - dy * dy);
#pragma unroll
        for (int c = 9; c < 16; ++c) s[c] = 0.0f;
        v8h h0, h1v;
#pragma unroll
        for (int c = 0; c < 8; ++c) { h0[c] = (_Float16)s[c]; h1v[c] = (_Float16)s[c + 8]; }
        *(v8h*)&sSH[wave][m * 32]     = h0;
        *(v8h*)&sSH[wave][m * 32 + 8] = h1v;
    } else {
        v8h z = {};
        *(v8h*)&sSH[wave][m * 32 + 16] = z;
        *(v8h*)&sSH[wave][m * 32 + 24] = z;
    }

    wave_lds_fence();

    // ---- Layer 1: base_x = pc @ W_base^T + b_base ; color_pre = base_x + sh @ W_dir^T + b_dir ----
    const v16h apc = load_frag(&sA[wave][m * 32], g0);
    const v16h ash = load_frag(&sSH[wave][m * 32], g0);

#pragma unroll
    for (int t = 0; t < 4; ++t) {
        const int n = t * 16 + m;          // output column owned by this lane
        v16h bw = load_frag(&sWB[n * 32], g0);
        v8f zero = {};
        v8f accb = __builtin_amdgcn_wmma_f32_16x16x32_f16(
            false, apc, false, bw, (short)0, zero, false, false);
        const float bb = b_base[n];
#pragma unroll
        for (int r = 0; r < 8; ++r) accb[r] += bb;   // base_x

        v16h bd = load_frag(&sWD[n * 32], g0);
        v8f accc = __builtin_amdgcn_wmma_f32_16x16x32_f16(
            false, ash, false, bd, (short)0, accb, false, false);
        const float bdv = b_dir[n];

#pragma unroll
        for (int r = 0; r < 8; ++r) {
            const float a1 = fast_silu(accb[r]);          // base_act
            const float a2 = fast_silu(accc[r] + bdv);    // color_in (activated)
            const int mm = r + hi * 8;                    // C/D row -> point index in tile
            sAct1[wave][mm * 64 + n] = (_Float16)a1;
            sAct2[wave][mm * 64 + n] = (_Float16)a2;
        }
    }

    wave_lds_fence();

    // ---- Heads: K=64 as two K=32 WMMAs each ----
    const v16h a1lo = load_frag(&sAct1[wave][m * 64],      g0);
    const v16h a1hi = load_frag(&sAct1[wave][m * 64 + 32], g0);
    const v16h a2lo = load_frag(&sAct2[wave][m * 64],      g0);
    const v16h a2hi = load_frag(&sAct2[wave][m * 64 + 32], g0);
    const v16h b1lo = load_frag(&sB1[m * 64],      g0);
    const v16h b1hi = load_frag(&sB1[m * 64 + 32], g0);
    const v16h b2lo = load_frag(&sB2[m * 64],      g0);
    const v16h b2hi = load_frag(&sB2[m * 64 + 32], g0);

    v8f zero = {};
    v8f h1 = __builtin_amdgcn_wmma_f32_16x16x32_f16(false, a1lo, false, b1lo, (short)0, zero, false, false);
    h1     = __builtin_amdgcn_wmma_f32_16x16x32_f16(false, a1hi, false, b1hi, (short)0, h1,   false, false);
    v8f h2 = __builtin_amdgcn_wmma_f32_16x16x32_f16(false, a2lo, false, b2lo, (short)0, zero, false, false);
    h2     = __builtin_amdgcn_wmma_f32_16x16x32_f16(false, a2hi, false, b2hi, (short)0, h2,   false, false);

    // ---- Epilogue: lane column m selects output role ----
    if (m == 0) {
        const float bden = b_density[0];
#pragma unroll
        for (int r = 0; r < 8; ++r) {
            const int point = p0 + r + hi * 8;
            out[point] = __expf(h1[r] + bden);                 // sigma
        }
    } else if (m <= 3) {
        const int c = m - 1;
        const float bc = b_color[c];
        const float bw_ = b_color_wodir[c];
#pragma unroll
        for (int r = 0; r < 8; ++r) {
            const int point = p0 + r + hi * 8;
            const float cg = fast_sigmoid(h2[r] + bc);         // dir color
            const float cn = fast_sigmoid(h1[r] + bw_);        // no-dir color
            const float v = (0.8f * cg + 0.2f * cn) * 1.002f - 0.001f;
            out[npts + point * 3 + c] = v;
        }
    }
}

extern "C" void kernel_launch(void* const* d_in, const int* in_sizes, int n_in,
                              void* d_out, int out_size, void* d_ws, size_t ws_size,
                              hipStream_t stream) {
    const float* code          = (const float*)d_in[0];
    const float* pts           = (const float*)d_in[1];
    const float* dirs          = (const float*)d_in[2];
    const float* w_base        = (const float*)d_in[3];
    const float* b_base        = (const float*)d_in[4];
    const float* w_density     = (const float*)d_in[5];
    const float* b_density     = (const float*)d_in[6];
    const float* w_dir         = (const float*)d_in[7];
    const float* b_dir         = (const float*)d_in[8];
    const float* w_color       = (const float*)d_in[9];
    const float* b_color       = (const float*)d_in[10];
    const float* w_color_wodir = (const float*)d_in[11];
    const float* b_color_wodir = (const float*)d_in[12];
    float* out = (float*)d_out;

    const int npts  = in_sizes[2] / 3;       // dirs is (N,3)
    const int tiles = (npts + 15) / 16;
    const int blocks = (tiles + 7) / 8;      // 8 waves (tiles) per 256-thread block

    tensorial_decoder_kernel<<<blocks, 256, 0, stream>>>(
        code, pts, dirs, w_base, b_base, w_density, b_density,
        w_dir, b_dir, w_color, b_color, w_color_wodir, b_color_wodir,
        out, npts);
}
